// GNNDecoder_37349035606505
// MI455X (gfx1250) — matrix operand order repaired and checked by reading, not compiled, not run on hardware
//
#include <hip/hip_runtime.h>
#include <stddef.h>

// ---------------------------------------------------------------------------
// Types for CDNA5 WMMA (wave32): D = A(16x32 bf16) x B(32x16 bf16) + C(16x16 f32)
// ---------------------------------------------------------------------------
typedef __attribute__((ext_vector_type(16))) __bf16 v16bf;
typedef __attribute__((ext_vector_type(8)))  float  v8f;

union BF16Frag {
    v16bf    v;
    unsigned u[8];
    uint4    q[2];
};

__device__ __forceinline__ unsigned short f2bf(float f) {
    unsigned u = __builtin_bit_cast(unsigned, f);
    u += 0x7FFFu + ((u >> 16) & 1u);        // round-to-nearest-even
    return (unsigned short)(u >> 16);
}
__device__ __forceinline__ unsigned packbf(float lo, float hi) {
    return (unsigned)f2bf(lo) | ((unsigned)f2bf(hi) << 16);
}
__device__ __forceinline__ v8f wmma_bf16(const BF16Frag& a, const BF16Frag& b, v8f c) {
    return __builtin_amdgcn_wmma_f32_16x16x32_bf16(
        /*neg_a=*/false, a.v, /*neg_b=*/false, b.v,
        /*c_mod=*/(short)0, c, /*reuse_a=*/false, /*reuse_b=*/false);
}

// XOR swizzle on dword-column bits [5:2]: keeps 4-dword blocks contiguous and
// 16B-aligned (ds_load_b128) while 16 consecutive rows hit 16 distinct banks.
__device__ __forceinline__ int swz(int row, int c) {
    return c ^ ((row & 15) << 2);
}

__device__ __forceinline__ unsigned lds_off(const void* p) {
    return (unsigned)(size_t)p;             // low 32 bits of generic ptr = LDS offset
}

// Async copy of one 16B chunk: global (saddr + 32-bit voffset) -> LDS, ASYNCcnt.
__device__ __forceinline__ void async_copy16(unsigned lds_byte, unsigned gl_byte,
                                             const void* base) {
    asm volatile("global_load_async_to_lds_b128 %0, %1, %2"
                 :: "v"(lds_byte), "v"(gl_byte), "s"(base) : "memory");
}
__device__ __forceinline__ void async_wait0() {
    asm volatile("s_wait_asynccnt 0x0" ::: "memory");
}

#define DIM 128

// ---------------------------------------------------------------------------
// K0: pack W_lin (128x128), W1 (256x128), W2 (128x256) from f32 -> bf16 pairs
// ---------------------------------------------------------------------------
__global__ __launch_bounds__(256) void pack_weights(
    const float* __restrict__ Wlin, const float* __restrict__ W1,
    const float* __restrict__ W2,
    unsigned* __restrict__ wlinbf, unsigned* __restrict__ w1bf,
    unsigned* __restrict__ w2bf) {
    int i = blockIdx.x * 256 + threadIdx.x;     // dword (pair) index
    if (i < 8192) {
        wlinbf[i] = packbf(Wlin[2 * i], Wlin[2 * i + 1]);
    } else if (i < 8192 + 16384) {
        int j = i - 8192;
        w1bf[j] = packbf(W1[2 * j], W1[2 * j + 1]);
    } else if (i < 8192 + 16384 + 16384) {
        int j = i - 24576;
        w2bf[j] = packbf(W2[2 * j], W2[2 * j + 1]);
    }
}

// ---------------------------------------------------------------------------
// K1: h = PReLU(x) @ W_lin^T    (M=N, K=128, N=128), h written to d_out (temp)
// Block = 128 threads = 4 waves; each wave owns one 16-row tile x all 128 cols.
// ---------------------------------------------------------------------------
__global__ __launch_bounds__(128) void gemm1_prelu(
    const float* __restrict__ x, const float* __restrict__ prelu_w,
    const unsigned* __restrict__ wlinbf, float* __restrict__ h, int nrows) {
    __shared__ __align__(16) unsigned sW[128 * 64];     // 32 KB, swizzled bf16 pairs
    __shared__ __align__(16) unsigned sA[4][16 * 64];   // 16 KB

    const int tid = threadIdx.x, lane = tid & 31, wave = tid >> 5;
    const float pw = prelu_w[0];

    // async-stage W_lin: 128 rows x 16 chunks of 16B
    {
        unsigned base = lds_off(&sW[0]);
        for (int i = tid; i < 128 * 16; i += 128) {
            int r = i >> 4, c4 = (i & 15) << 2;
            unsigned dst = base + (unsigned)((r * 64 + swz(r, c4)) * 4);
            async_copy16(dst, (unsigned)(i * 16), wlinbf);
        }
    }

    const int ntiles = (nrows + 15) >> 4;
    const int tile   = blockIdx.x * 4 + wave;
    const bool active = tile < ntiles;

    if (active) {
        unsigned* sa = sA[wave];
        for (int i = lane; i < 16 * 64; i += 32) {
            int r = i >> 6, c = i & 63;
            int gr = tile * 16 + r;
            float f0 = 0.f, f1 = 0.f;
            if (gr < nrows) {
                f0 = x[(size_t)gr * DIM + 2 * c];
                f1 = x[(size_t)gr * DIM + 2 * c + 1];
                f0 = f0 >= 0.f ? f0 : pw * f0;
                f1 = f1 >= 0.f ? f1 : pw * f1;
            }
            sa[r * 64 + swz(r, c)] = packbf(f0, f1);
        }
    }
    async_wait0();
    __syncthreads();
    if (!active) return;

    const int lm = lane & 15, lg = lane >> 4;
    unsigned* sa = sA[wave];

    v8f acc[8] = {};
#pragma unroll
    for (int kc = 0; kc < 4; ++kc) {            // K chunks of 32
        BF16Frag a;
        a.q[0] = *(const uint4*)&sa[lm * 64 + swz(lm, kc * 16 + lg * 4)];
        a.q[1] = *(const uint4*)&sa[lm * 64 + swz(lm, kc * 16 + lg * 4 + 8)];
#pragma unroll
        for (int nt = 0; nt < 8; ++nt) {
            int o = nt * 16 + lm;               // output col for this lane
            BF16Frag b;
            b.q[0] = *(const uint4*)&sW[o * 64 + swz(o, kc * 16 + lg * 4)];
            b.q[1] = *(const uint4*)&sW[o * 64 + swz(o, kc * 16 + lg * 4 + 8)];
            acc[nt] = wmma_bf16(a, b, acc[nt]);
        }
    }
    // C layout: VGPR r -> (M = r + lg*8, N = nt*16 + lm)
#pragma unroll
    for (int nt = 0; nt < 8; ++nt) {
#pragma unroll
        for (int r = 0; r < 8; ++r) {
            int gr = tile * 16 + r + lg * 8;
            if (gr < nrows) h[(size_t)gr * DIM + nt * 16 + lm] = acc[nt][r];
        }
    }
}

// ---------------------------------------------------------------------------
// K2a: aggr = h + edge_emb1[4] + edge_emb2[0]   (self-loop contribution)
// ---------------------------------------------------------------------------
__global__ __launch_bounds__(256) void init_aggr(
    const float* __restrict__ h, const float* __restrict__ eb1,
    const float* __restrict__ eb2, float* __restrict__ aggr, int total4) {
    int i = blockIdx.x * 256 + threadIdx.x;
    if (i >= total4) return;
    int c = (i & 31) * 4;
    float4 hv = ((const float4*)h)[i];
    float4 a  = *(const float4*)(eb1 + 4 * DIM + c);
    float4 b  = *(const float4*)(eb2 + 0 * DIM + c);
    float4 o;
    o.x = hv.x + a.x + b.x;  o.y = hv.y + a.y + b.y;
    o.z = hv.z + a.z + b.z;  o.w = hv.w + a.w + b.w;
    ((float4*)aggr)[i] = o;
}

// ---------------------------------------------------------------------------
// K2b: scatter-add messages. One wave per edge; lane handles 4 floats.
// aggr is L2-resident (51MB << 192MB L2) so atomics stay on-chip.
// ---------------------------------------------------------------------------
__global__ __launch_bounds__(256) void scatter_edges(
    const int* __restrict__ ei, const int* __restrict__ ea,
    const float* __restrict__ h, const float* __restrict__ eb1,
    const float* __restrict__ eb2, float* __restrict__ aggr, int E) {
    int lane = threadIdx.x & 31;
    // wave-uniform edge id -> scalar loads of edge metadata
    int e = __builtin_amdgcn_readfirstlane(blockIdx.x * 8 + (threadIdx.x >> 5));
    if (e >= E) return;
    int src = ei[e], dst = ei[E + e];
    int t0 = ea[2 * e], t1 = ea[2 * e + 1];
    int c = lane * 4;
    float4 hv = *(const float4*)(h + (size_t)src * DIM + c);
    float4 a  = *(const float4*)(eb1 + (size_t)t0 * DIM + c);
    float4 b  = *(const float4*)(eb2 + (size_t)t1 * DIM + c);
    float* p = aggr + (size_t)dst * DIM + c;
    atomicAdd(p + 0, hv.x + a.x + b.x);
    atomicAdd(p + 1, hv.y + a.y + b.y);
    atomicAdd(p + 2, hv.z + a.z + b.z);
    atomicAdd(p + 3, hv.w + a.w + b.w);
}

// ---------------------------------------------------------------------------
// K3: fused MLP:  out = relu(aggr @ W1^T + b1) @ W2^T + b2
// Block = 64 threads = 2 waves, each wave owns a 16-row tile. hmid never
// leaves LDS (saves ~205 MB of HBM traffic). Static LDS = 155.6 KB.
// ---------------------------------------------------------------------------
__global__ __launch_bounds__(64) void mlp_fused(
    const float* __restrict__ aggr, const unsigned* __restrict__ w1bf,
    const unsigned* __restrict__ w2bf, const float* __restrict__ b1,
    const float* __restrict__ b2, float* __restrict__ out, int nrows) {
    __shared__ __align__(16) unsigned sW1[256 * 64];       // 64 KB  (K=128)
    __shared__ __align__(16) unsigned sW2[128 * 128];      // 64 KB  (K=256)
    __shared__ __align__(16) unsigned sA [2][16 * 64];     //  8 KB
    __shared__ __align__(16) unsigned sH [2][16 * 128];    // 16 KB  (hmid bf16)

    const int tid = threadIdx.x, lane = tid & 31, wave = tid >> 5;

    // async-stage W1 (256 rows x 16 chunks) and W2 (128 rows x 32 chunks)
    {
        unsigned b1off = lds_off(&sW1[0]);
        for (int i = tid; i < 256 * 16; i += 64) {
            int r = i >> 4, c4 = (i & 15) << 2;
            async_copy16(b1off + (unsigned)((r * 64 + swz(r, c4)) * 4),
                         (unsigned)(i * 16), w1bf);
        }
        unsigned b2off = lds_off(&sW2[0]);
        for (int i = tid; i < 128 * 32; i += 64) {
            int r = i >> 5, c4 = (i & 31) << 2;
            async_copy16(b2off + (unsigned)((r * 128 + swz(r, c4)) * 4),
                         (unsigned)(i * 16), w2bf);
        }
    }

    const int ntiles = (nrows + 15) >> 4;
    const int tile   = blockIdx.x * 2 + wave;
    const bool active = tile < ntiles;

    if (active) {
        unsigned* sa = sA[wave];
        for (int i = lane; i < 16 * 64; i += 32) {
            int r = i >> 6, c = i & 63;
            int gr = tile * 16 + r;
            float f0 = 0.f, f1 = 0.f;
            if (gr < nrows) {
                f0 = aggr[(size_t)gr * DIM + 2 * c];
                f1 = aggr[(size_t)gr * DIM + 2 * c + 1];
            }
            sa[r * 64 + swz(r, c)] = packbf(f0, f1);
        }
    }
    async_wait0();
    __syncthreads();

    const int lm = lane & 15, lg = lane >> 4;

    if (active) {
        unsigned* sa = sA[wave];
        unsigned short* shs = (unsigned short*)sH[wave];
        // ---- stage 1: hmid = relu(aggr @ W1^T + b1), 256 cols in two halves
#pragma unroll
        for (int half = 0; half < 2; ++half) {
            v8f acc[8] = {};
#pragma unroll
            for (int kc = 0; kc < 4; ++kc) {
                BF16Frag a;
                a.q[0] = *(const uint4*)&sa[lm * 64 + swz(lm, kc * 16 + lg * 4)];
                a.q[1] = *(const uint4*)&sa[lm * 64 + swz(lm, kc * 16 + lg * 4 + 8)];
#pragma unroll
                for (int nt = 0; nt < 8; ++nt) {
                    int o = half * 128 + nt * 16 + lm;   // W1 output row
                    BF16Frag b;
                    b.q[0] = *(const uint4*)&sW1[o * 64 + swz(o, kc * 16 + lg * 4)];
                    b.q[1] = *(const uint4*)&sW1[o * 64 + swz(o, kc * 16 + lg * 4 + 8)];
                    acc[nt] = wmma_bf16(a, b, acc[nt]);
                }
            }
#pragma unroll
            for (int nt = 0; nt < 8; ++nt) {
                int col = half * 128 + nt * 16 + lm;
                float bias = b1[col];
#pragma unroll
                for (int r = 0; r < 8; ++r) {
                    float f = acc[nt][r] + bias;
                    f = f > 0.f ? f : 0.f;
                    int m = r + lg * 8;
                    int d = m * 128 + swz(m, col >> 1);   // dword in sH row (pitch 128)
                    shs[2 * d + (col & 1)] = f2bf(f);
                }
            }
        }
    }
    __syncthreads();

    if (active) {
        unsigned* shd = sH[wave];
        // ---- stage 2: out = hmid @ W2^T + b2   (K = 256)
        v8f acc[8] = {};
#pragma unroll
        for (int kc = 0; kc < 8; ++kc) {
            BF16Frag a;
            a.q[0] = *(const uint4*)&shd[lm * 128 + swz(lm, kc * 16 + lg * 4)];
            a.q[1] = *(const uint4*)&shd[lm * 128 + swz(lm, kc * 16 + lg * 4 + 8)];
#pragma unroll
            for (int nt = 0; nt < 8; ++nt) {
                int o = nt * 16 + lm;                  // W2 output row
                BF16Frag b;
                b.q[0] = *(const uint4*)&sW2[o * 128 + swz(o, kc * 16 + lg * 4)];
                b.q[1] = *(const uint4*)&sW2[o * 128 + swz(o, kc * 16 + lg * 4 + 8)];
                acc[nt] = wmma_bf16(a, b, acc[nt]);
            }
        }
#pragma unroll
        for (int nt = 0; nt < 8; ++nt) {
            int col = nt * 16 + lm;
            float bias = b2[col];
#pragma unroll
            for (int r = 0; r < 8; ++r) {
                int gr = tile * 16 + r + lg * 8;
                if (gr < nrows) out[(size_t)gr * DIM + col] = acc[nt][r] + bias;
            }
        }
    }
}

// ---------------------------------------------------------------------------
// Launch
// ---------------------------------------------------------------------------
extern "C" void kernel_launch(void* const* d_in, const int* in_sizes, int n_in,
                              void* d_out, int out_size, void* d_ws, size_t ws_size,
                              hipStream_t stream) {
    const float* x       = (const float*)d_in[0];
    const float* prelu_w = (const float*)d_in[1];
    const float* W_lin   = (const float*)d_in[2];
    const float* eemb1   = (const float*)d_in[3];
    const float* eemb2   = (const float*)d_in[4];
    const float* W1      = (const float*)d_in[5];
    const float* b1      = (const float*)d_in[6];
    const float* W2      = (const float*)d_in[7];
    const float* b2      = (const float*)d_in[8];
    const int*   ei      = (const int*)d_in[9];
    const int*   ea      = (const int*)d_in[10];

    const int N = in_sizes[0] / DIM;     // 100000
    const int E = in_sizes[9] / 2;       // 1600000

    // workspace layout
    char* ws = (char*)d_ws;
    float*    aggr   = (float*)ws;                                   // N*128 f32
    size_t    off    = (size_t)N * DIM * sizeof(float);
    unsigned* wlinbf = (unsigned*)(ws + off);            off += 128 * 64 * 4;
    unsigned* w1bf   = (unsigned*)(ws + off);            off += 256 * 64 * 4;
    unsigned* w2bf   = (unsigned*)(ws + off);

    float* h   = (float*)d_out;   // d_out doubles as scratch for h
    float* out = (float*)d_out;

    // K0: pack weights to bf16
    pack_weights<<<(40960 + 255) / 256, 256, 0, stream>>>(W_lin, W1, W2,
                                                          wlinbf, w1bf, w2bf);
    // K1: h = PReLU(x) @ W_lin^T
    {
        int tiles  = (N + 15) / 16;
        int blocks = (tiles + 3) / 4;
        gemm1_prelu<<<blocks, 128, 0, stream>>>(x, prelu_w, wlinbf, h, N);
    }
    // K2a: aggr = h + self-loop embedding
    {
        int total4 = N * (DIM / 4);
        init_aggr<<<(total4 + 255) / 256, 256, 0, stream>>>(h, eemb1, eemb2,
                                                            aggr, total4);
    }
    // K2b: scatter edges (one wave per edge)
    {
        int blocks = (E + 7) / 8;
        scatter_edges<<<blocks, 256, 0, stream>>>(ei, ea, h, eemb1, eemb2,
                                                  aggr, E);
    }
    // K3: fused GIN MLP
    {
        int tiles  = (N + 15) / 16;
        int blocks = (tiles + 1) / 2;
        mlp_fused<<<blocks, 64, 0, stream>>>(aggr, w1bf, w2bf, b1, b2, out, N);
    }
}
